// MultiHeadAttention_26147760898452
// MI455X (gfx1250) — matrix operand order repaired and checked by reading, not compiled, not run on hardware
//
#include <hip/hip_runtime.h>
#include <hip/hip_bf16.h>

typedef __bf16 bf16;
typedef __attribute__((ext_vector_type(4)))  bf16  v4bf;
typedef __attribute__((ext_vector_type(8)))  bf16  v8bf;
typedef __attribute__((ext_vector_type(16))) bf16  v16bf;
typedef __attribute__((ext_vector_type(8)))  float v8f;

#define NB 4
#define LL 2048
#define EE 1024
#define HH 16
#define DD 64
#define HD (HH*DD)

// ---- helpers -------------------------------------------------------------

// 16-bit A-operand layout (contiguous, matches B): lanes 0-15 hold K {0..15},
// lanes 16-31 hold K {16..31}.  p = row base + k base (32-K window).
static __device__ inline v16bf load_a_frag(const bf16* p, int half) {
  v8bf lo = *(const v8bf*)(p + half * 16);
  v8bf hi = *(const v8bf*)(p + half * 16 + 8);
  return __builtin_shufflevector(lo, hi, 0,1,2,3,4,5,6,7,8,9,10,11,12,13,14,15);
}

static __device__ inline v8f wmma_bf16(v16bf a, v16bf b, v8f c) {
  return __builtin_amdgcn_wmma_f32_16x16x32_bf16(false, a, false, b,
                                                 (short)0, c, false, false);
}

// async global -> LDS, 16 bytes per lane (tracked with ASYNCcnt)
static __device__ inline void async_b128(const bf16* gptr, void* lptr) {
  unsigned laddr = (unsigned)(size_t)lptr;             // low 32 bits = LDS offset
  unsigned long long gaddr = (unsigned long long)(size_t)gptr;
  asm volatile("global_load_async_to_lds_b128 %0, %1, off"
               :: "v"(laddr), "v"(gaddr) : "memory");
}

// ---- fp32 -> bf16 convert (float4 vectorized) ----------------------------

__global__ void cvt_f32_bf16(const float* __restrict__ in,
                             bf16* __restrict__ out, int n4) {
  const int i = blockIdx.x * blockDim.x + threadIdx.x;
  if (i >= n4) return;
  const float4 f = ((const float4*)in)[i];
  v4bf o;
  o[0] = (bf16)f.x; o[1] = (bf16)f.y; o[2] = (bf16)f.z; o[3] = (bf16)f.w;
  *(v4bf*)(out + (size_t)i * 4) = o;
}

// ---- vT[n,h,d,l] = vp[n,l,h,d] -------------------------------------------

__global__ void transpose_vhd(const bf16* __restrict__ vp,
                              bf16* __restrict__ vTo) {
  const int i = blockIdx.x * blockDim.x + threadIdx.x;   // NB*HH*DD*LL
  const int l = i & (LL - 1);
  int t = i >> 11;
  const int d = t & (DD - 1); t >>= 6;
  const int h = t & (HH - 1);
  const int n = t >> 4;
  vTo[i] = vp[((size_t)(n * LL + l) * HH + h) * DD + d];
}

// ---- GEMM: out[M x Nn] = A[M x K] * B[Nn x K]^T + bias -------------------
// Wave tile 16M x 64N, 4 waves/block along M (block tile 64M x 64N).
// B tile (64N x 32K) staged in LDS via async global->LDS, double buffered.
// OUTF32: 1 -> f32 out, 0 -> bf16 out.

template <int OUTF32>
__global__ __launch_bounds__(128) void gemm_wmma(
    const bf16* __restrict__ A, const bf16* __restrict__ B,
    const float* __restrict__ bias, void* __restrict__ out,
    int M, int Nn, int Kd) {
  __shared__ __align__(64) bf16 sB[2][64][32];   // 8 KB double buffer

  const int t    = threadIdx.x;
  const int lane = t & 31;
  const int w    = t >> 5;
  const int half = lane >> 4;
  const int ln   = lane & 15;
  const int m0 = blockIdx.y * 64 + w * 16;
  const int n0 = blockIdx.x * 64;

  const bf16* arow = A + (size_t)(m0 + ln) * Kd;

  // stage B[n0..n0+64) x [k..k+32) into sB[buf]: 4 KB = 128 thr x 2 x 16 B
  auto stage = [&](int buf, int k) {
#pragma unroll
    for (int i = 0; i < 2; ++i) {
      const int chunk = t + i * 128;          // 0..255
      const int row   = chunk >> 2;           // 0..63
      const int koff  = (chunk & 3) * 8;      // element offset in 32-K tile
      async_b128(B + (size_t)(n0 + row) * Kd + k + koff, &sB[buf][row][koff]);
    }
  };

  v8f acc[4] = {};
  const int nk = Kd / 32;
  stage(0, 0);
  for (int ki = 0; ki < nk; ++ki) {
    asm volatile("s_wait_asynccnt 0x0" ::: "memory");  // my tile-ki writes done
    __syncthreads();                                   // everyone's writes visible
    if (ki + 1 < nk) stage((ki + 1) & 1, (ki + 1) * 32);

    v16bf a = load_a_frag(arow + ki * 32, half);
    __builtin_prefetch(arow + ki * 32 + 256, 0, 0);
#pragma unroll
    for (int c = 0; c < 4; ++c) {
      v16bf b = *(const v16bf*)&sB[ki & 1][c * 16 + ln][half * 16];
      acc[c] = wmma_bf16(a, b, acc[c]);
    }
    // no trailing barrier needed: iteration ki+1 stages into buffer (ki+2)&1,
    // whose readers all passed the barrier at the top of iteration ki+1.
  }

#pragma unroll
  for (int c = 0; c < 4; ++c) {
    const int col  = n0 + c * 16 + ln;
    const float bb = bias[col];
#pragma unroll
    for (int j = 0; j < 8; ++j) {
      const int row = m0 + j + 8 * half;
      const float o = acc[c][j] + bb;
      if (OUTF32) ((float*)out)[(size_t)row * Nn + col] = o;
      else        ((bf16*)out)[(size_t)row * Nn + col] = (bf16)o;
    }
  }
}

// ---- Flash attention: one wave per (n, h, 16-query tile) -----------------

__global__ __launch_bounds__(128) void attn_flash(
    const bf16* __restrict__ qp, const bf16* __restrict__ kp,
    const bf16* __restrict__ vT, bf16* __restrict__ ctx) {
  __shared__ __align__(64) bf16 sP[4][16][32];

  const int lane = threadIdx.x & 31;
  const int w    = threadIdx.x >> 5;
  const int half = lane >> 4;
  const int ln   = lane & 15;
  const int tile = blockIdx.x * 4 + w;          // 8192 tiles
  const int n  = tile >> 11;
  const int h  = (tile >> 7) & (HH - 1);
  const int qt = tile & 127;

  // loop-invariant q A-frags (d = K, 64 -> two 32-K windows)
  const bf16* qrow = qp + ((size_t)(n * LL + qt * 16 + ln) * HH + h) * DD;
  const v16bf a0 = load_a_frag(qrow, half);
  const v16bf a1 = load_a_frag(qrow + 32, half);

  v8f acc[4] = {};
  float m[8], lsum[8];
#pragma unroll
  for (int j = 0; j < 8; ++j) { m[j] = -1e30f; lsum[j] = 0.0f; }

  const bf16* kb_base = kp + ((size_t)(n * LL + ln) * HH + h) * DD + half * 16;
  const bf16* vt_base = vT + ((size_t)(n * HH + h) * DD + ln) * LL + half * 16;

  for (int kb = 0; kb < LL; kb += 32) {
    // S = (q @ k^T) for keys [kb, kb+32)
    const bf16* k0 = kb_base + (size_t)kb * HD;
    const bf16* k1 = k0 + (size_t)16 * HD;
    v8f s0 = {}, s1 = {};
    s0 = wmma_bf16(a0, *(const v16bf*)(k0),      s0);
    s0 = wmma_bf16(a1, *(const v16bf*)(k0 + 32), s0);
    s1 = wmma_bf16(a0, *(const v16bf*)(k1),      s1);
    s1 = wmma_bf16(a1, *(const v16bf*)(k1 + 32), s1);

    // online softmax; C-layout row = j + 8*half, col = ln (+16 for s1)
#pragma unroll
    for (int j = 0; j < 8; ++j) {
      float x0 = s0[j] * 0.125f;                 // 1/sqrt(D)
      float x1 = s1[j] * 0.125f;
      float mx = fmaxf(x0, x1);
#pragma unroll
      for (int off = 1; off < 16; off <<= 1)
        mx = fmaxf(mx, __shfl_xor(mx, off, 32));
      const float mn = fmaxf(m[j], mx);
      const float sc = __expf(m[j] - mn);
      m[j] = mn;
      const float p0 = __expf(x0 - mn);
      const float p1 = __expf(x1 - mn);
      float r = p0 + p1;
#pragma unroll
      for (int off = 1; off < 16; off <<= 1)
        r += __shfl_xor(r, off, 32);
      lsum[j] = lsum[j] * sc + r;
#pragma unroll
      for (int c = 0; c < 4; ++c) acc[c][j] *= sc;
      const int row = j + 8 * half;
      sP[w][row][ln]      = (bf16)p0;
      sP[w][row][ln + 16] = (bf16)p1;
    }

    // intra-wave LDS RAW fence (cross-lane, so force the wait explicitly)
    asm volatile("s_wait_dscnt 0x0" ::: "memory");

    // P (16x32, A-operand from LDS) @ V (32 keys x 64 d, B from vT)
    v16bf pa = load_a_frag(&sP[w][ln][0], half);
#pragma unroll
    for (int c = 0; c < 4; ++c) {
      v16bf bfrag = *(const v16bf*)(vt_base + (size_t)c * 16 * LL + kb);
      acc[c] = wmma_bf16(pa, bfrag, acc[c]);
    }
  }

  // normalize and scatter to ctx[n, l, h*D + d] (bf16)
  bf16* crow = ctx + (size_t)(n * LL + qt * 16) * HD + h * DD;
#pragma unroll
  for (int j = 0; j < 8; ++j) {
    const float inv = 1.0f / lsum[j];
    const int row = j + 8 * half;
#pragma unroll
    for (int c = 0; c < 4; ++c)
      crow[(size_t)row * HD + c * 16 + ln] = (bf16)(acc[c][j] * inv);
  }
}

// ---- launch --------------------------------------------------------------

extern "C" void kernel_launch(void* const* d_in, const int* in_sizes, int n_in,
                              void* d_out, int out_size, void* d_ws,
                              size_t ws_size, hipStream_t stream) {
  (void)in_sizes; (void)n_in; (void)out_size; (void)ws_size;
  const float* Q   = (const float*)d_in[0];
  const float* K   = (const float*)d_in[1];
  const float* V   = (const float*)d_in[2];
  const float* Wq  = (const float*)d_in[3];
  const float* bq  = (const float*)d_in[4];
  const float* Wk  = (const float*)d_in[5];
  const float* bk  = (const float*)d_in[6];
  const float* Wv  = (const float*)d_in[7];
  const float* bvv = (const float*)d_in[8];
  const float* Wo  = (const float*)d_in[9];
  const float* bo  = (const float*)d_in[10];

  const size_t XE = (size_t)NB * LL * EE;  // 8388608
  const size_t WE = (size_t)HD * EE;       // 1048576

  char* p = (char*)d_ws;
  bf16* Qb  = (bf16*)p; p += XE * sizeof(bf16);
  bf16* Kb  = (bf16*)p; p += XE * sizeof(bf16);
  bf16* Vb  = (bf16*)p; p += XE * sizeof(bf16);
  bf16* Wqb = (bf16*)p; p += WE * sizeof(bf16);
  bf16* Wkb = (bf16*)p; p += WE * sizeof(bf16);
  bf16* Wvb = (bf16*)p; p += WE * sizeof(bf16);
  bf16* Wob = (bf16*)p; p += WE * sizeof(bf16);
  bf16* qp  = (bf16*)p; p += XE * sizeof(bf16);
  bf16* kp  = (bf16*)p; p += XE * sizeof(bf16);
  bf16* vp  = (bf16*)p; p += XE * sizeof(bf16);
  bf16* vT  = (bf16*)p; p += XE * sizeof(bf16);
  bf16* ctx = (bf16*)p; p += XE * sizeof(bf16);

  // fp32 -> bf16
  cvt_f32_bf16<<<(int)(XE / 4 / 256), 256, 0, stream>>>(Q, Qb, (int)(XE / 4));
  cvt_f32_bf16<<<(int)(XE / 4 / 256), 256, 0, stream>>>(K, Kb, (int)(XE / 4));
  cvt_f32_bf16<<<(int)(XE / 4 / 256), 256, 0, stream>>>(V, Vb, (int)(XE / 4));
  cvt_f32_bf16<<<(int)(WE / 4 / 256), 256, 0, stream>>>(Wq, Wqb, (int)(WE / 4));
  cvt_f32_bf16<<<(int)(WE / 4 / 256), 256, 0, stream>>>(Wk, Wkb, (int)(WE / 4));
  cvt_f32_bf16<<<(int)(WE / 4 / 256), 256, 0, stream>>>(Wv, Wvb, (int)(WE / 4));
  cvt_f32_bf16<<<(int)(WE / 4 / 256), 256, 0, stream>>>(Wo, Wob, (int)(WE / 4));

  // projections (bf16 WMMA, f32 accum, bf16 out)
  dim3 ggrid(EE / 64, (NB * LL) / 64);   // 16 x 128
  gemm_wmma<0><<<ggrid, 128, 0, stream>>>(Qb, Wqb, bq,  (void*)qp, NB * LL, HD, EE);
  gemm_wmma<0><<<ggrid, 128, 0, stream>>>(Kb, Wkb, bk,  (void*)kp, NB * LL, HD, EE);
  gemm_wmma<0><<<ggrid, 128, 0, stream>>>(Vb, Wvb, bvv, (void*)vp, NB * LL, HD, EE);

  // v -> vT[n,h,d,l] for the P@V B-operand
  transpose_vhd<<<(int)(XE / 256), 256, 0, stream>>>(vp, vT);

  // flash attention
  attn_flash<<<(NB * HH * (LL / 16)) / 4, 128, 0, stream>>>(qp, kp, vT, ctx);

  // output projection, fp32 result straight into d_out
  dim3 ogrid(EE / 64, (NB * LL) / 64);
  gemm_wmma<1><<<ogrid, 128, 0, stream>>>(ctx, Wob, bo, d_out, NB * LL, EE, HD);
}
